// Res_up_13786845020771
// MI455X (gfx1250) — matrix operand order repaired and checked by reading, not compiled, not run on hardware
//
#include <hip/hip_runtime.h>

// ---------------------------------------------------------------------------
// Res_up graph-U-Net block for MI455X (gfx1250, wave32, WMMA).
//
// Every MLP is a fused kernel:
//   A(gather/concat/unpool, f32->f16) @ W1(f16 packed) -> +b1 -> ReLU
//   -> LDS (per-wave, row-major f16) -> @ W2(f16 packed) -> +b2
//   -> LayerNorm (shfl_xor row reduction) -> {store f32 | atomic scatter-sum
//      | residual+LeakyReLU}
// using v_wmma_f32_16x16x32_f16.  Segment widths are template parameters so
// per-K-chunk source selection folds at compile time; invalid (unpooled) rows
// read a shared zeroed row (no EXEC branching).  B fragments are loaded as a
// full per-chunk batch and A fragments are double-buffered so global-load
// latency overlaps the WMMA stream (explicit software pipelining).
// ---------------------------------------------------------------------------

typedef __attribute__((ext_vector_type(16))) _Float16 v16h;
typedef __attribute__((ext_vector_type(8)))  float    v8f;

#define N_LO 10000
#define N_HI 20000
#define E_LO 160000
#define E_HI 320000

// ---------------------------------------------------------------------------
// Weight packing: f32 row-major W[K][N]  ->  f16 B-fragment tiles.
// Tile (kt,nt) covers K=[kt*32,kt*32+32), N=[nt*16,nt*16+16).
// Lane l holds column n = nt*16 + (l&15); K half-range selected by l>=16.
// Half h of the lane's 16 halfs: k = kt*32 + (h<8 ? ((l>>4)&1)*8 + h : 16 + ((l>>4)&1)*8 + h-8).
// Packed offset = (((nt*KT + kt)*32) + lane)*16 + h   (halfs).
// ---------------------------------------------------------------------------
__global__ void pack_w_kernel(const float* __restrict__ W, _Float16* __restrict__ out,
                              int K, int N) {
  int idx = blockIdx.x * blockDim.x + threadIdx.x;
  if (idx >= K * N) return;
  int h    = idx & 15;
  int lane = (idx >> 4) & 31;
  int tile = idx >> 9;
  int KT   = K >> 5;
  int kt   = tile % KT;
  int nt   = tile / KT;
  int k = kt * 32 + ((lane >> 4) & 1) * 8 + (h & 7) + (h >> 3) * 16;
  int n = nt * 16 + (lane & 15);
  out[idx] = (_Float16)W[(size_t)k * N + n];
}

__global__ void scatter_ids_kernel(const int* __restrict__ ids, int n,
                                   int* __restrict__ inv) {
  int i = blockIdx.x * blockDim.x + threadIdx.x;
  if (i < n) inv[ids[i]] = i;
}

// ---------------------------------------------------------------------------
// A-fragment: 16 f16 halfs per lane.  Lane l: row m = l&15,
// kb = (l<16)?0:8 ; halfs 0..7 -> K = kb+0..7, halfs 8..15 -> K = kb+16..23
// (ISA 7.12.2 16-bit A 16x32 layout).  Pointer is always valid (zero-row for
// unpooled/missing rows) so the load is unconditional.
// ---------------------------------------------------------------------------
__device__ __forceinline__ v16h load_a_f32(const float* p) {
  float4 f0 = *(const float4*)(p + 0);
  float4 f1 = *(const float4*)(p + 4);
  float4 f2 = *(const float4*)(p + 16);
  float4 f3 = *(const float4*)(p + 20);
  v16h a;
  a[0]  = (_Float16)f0.x; a[1]  = (_Float16)f0.y; a[2]  = (_Float16)f0.z; a[3]  = (_Float16)f0.w;
  a[4]  = (_Float16)f1.x; a[5]  = (_Float16)f1.y; a[6]  = (_Float16)f1.z; a[7]  = (_Float16)f1.w;
  a[8]  = (_Float16)f2.x; a[9]  = (_Float16)f2.y; a[10] = (_Float16)f2.z; a[11] = (_Float16)f2.w;
  a[12] = (_Float16)f3.x; a[13] = (_Float16)f3.y; a[14] = (_Float16)f3.z; a[15] = (_Float16)f3.w;
  return a;
}

// ---------------------------------------------------------------------------
// Fused MLP (Linear-ReLU-Linear-LayerNorm) with fused gather (up to 3
// concatenated input segments, widths W0/W1/W2 as template params, optional
// 2-level indirection; -1 / out-of-range => zero row == unpooling) and fused
// epilogue (optional f32 store, optional scatter-sum via f32 atomics,
// optional residual + LeakyReLU).
// Block = 128 threads = 4 waves; wave w owns rows [blk*64+w*16, +16).
// ---------------------------------------------------------------------------
template <int W0, int W1, int W2, int DOUT>
__global__ __launch_bounds__(128) void mlp_fused(
    int nrows, const float* __restrict__ zrow,
    const float* __restrict__ s0, const int* __restrict__ s0i0, const int* __restrict__ s0i1,
    const float* __restrict__ s1, const int* __restrict__ s1i0, const int* __restrict__ s1i1,
    const float* __restrict__ s2, const int* __restrict__ s2i0, const int* __restrict__ s2i1,
    const _Float16* __restrict__ w1p, const float* __restrict__ b1,
    const _Float16* __restrict__ w2p, const float* __restrict__ b2,
    const float* __restrict__ ln_g, const float* __restrict__ ln_b,
    float* __restrict__ out, const int* __restrict__ scat, float* __restrict__ agg,
    const float* __restrict__ resid) {
  constexpr int DIN = W0 + W1 + W2;
  constexpr int KT1 = DIN / 32;
  constexpr int NT1 = DOUT / 16;
  constexpr int KT2 = DOUT / 32;
  constexpr int LSTR = DOUT + 8;  // f16 LDS row stride (keeps 16B alignment)

  __shared__ alignas(16) _Float16 h1[4][16 * LSTR];

  const int tid   = threadIdx.x;
  const int lane  = tid & 31;
  const int wave  = tid >> 5;
  const int row0  = blockIdx.x * 64 + wave * 16;
  const int arow  = row0 + (lane & 15);  // row this lane feeds into A fragments
  const int kb    = (lane & 16) ? 8 : 0;
  const int colb  = lane & 15;
  const int rhalf = (lane >> 4) & 1;

  // Resolve per-segment row base pointers once.  Invalid rows -> zrow, so the
  // inner-loop loads are unconditional (no EXEC-mask branching).
  auto seg_ptr = [&](const float* base, const int* i0, const int* i1,
                     int w) -> const float* {
    if (arow >= nrows) return zrow;
    int r = i0 ? i0[arow] : arow;
    if (r >= 0 && i1) r = i1[r];
    return (r >= 0) ? base + (size_t)r * w : zrow;
  };
  const float* p0 = seg_ptr(s0, s0i0, s0i1, W0);
  const float* p1 = seg_ptr(s1, s1i0, s1i1, W1);
  const float* p2 = (W2 > 0) ? seg_ptr(s2, s2i0, s2i1, W2) : zrow;

  // A-gather address for K-chunk c (selection folds after unroll).
  auto a_ptr = [&](int c) -> const float* {
    const int kc = c * 32;
    if (kc < W0)           return p0 + kc + kb;
    else if (kc < W0 + W1) return p1 + (kc - W0) + kb;
    else                   return p2 + (kc - W0 - W1) + kb;
  };

  // ---- layer 1: acc = A @ W1 (A double-buffered, B batched per chunk) ------
  v8f acc[NT1] = {};
  v16h a_cur = load_a_f32(a_ptr(0));
#pragma unroll
  for (int c = 0; c < KT1; ++c) {
    v16h a_next = (c + 1 < KT1) ? load_a_f32(a_ptr(c + 1)) : a_cur;
    v16h bf[NT1];
#pragma unroll
    for (int nt = 0; nt < NT1; ++nt)
      bf[nt] = *(const v16h*)(w1p + ((size_t)(nt * KT1 + c) * 32 + lane) * 16);
#pragma unroll
    for (int nt = 0; nt < NT1; ++nt)
      acc[nt] = __builtin_amdgcn_wmma_f32_16x16x32_f16(
          false, a_cur, false, bf[nt], (short)0, acc[nt], false, false);
    a_cur = a_next;
  }

  // ---- +b1, ReLU, f16 -> per-wave LDS (row-major) ---------------------------
#pragma unroll
  for (int nt = 0; nt < NT1; ++nt) {
    const int cg = nt * 16 + colb;
    const float bias = b1[cg];
#pragma unroll
    for (int v = 0; v < 8; ++v) {
      float t = acc[nt][v] + bias;
      t = t > 0.f ? t : 0.f;
      h1[wave][(rhalf * 8 + v) * LSTR + cg] = (_Float16)t;
    }
  }
  // LDS ops from one wave are in-order (DScnt); region is per-wave, so the
  // subsequent ds_loads see the stores without a workgroup barrier.

  // ---- layer 2: acc2 = H1 @ W2 (same pipelining; A now comes from LDS) -----
  v8f acc2[NT1] = {};
#pragma unroll
  for (int c = 0; c < KT2; ++c) {
    const _Float16* hp = &h1[wave][(lane & 15) * LSTR + c * 32 + kb];
    v16h a;
#pragma unroll
    for (int i = 0; i < 8; ++i) { a[i] = hp[i]; a[8 + i] = hp[16 + i]; }
    v16h bf[NT1];
#pragma unroll
    for (int nt = 0; nt < NT1; ++nt)
      bf[nt] = *(const v16h*)(w2p + ((size_t)(nt * KT2 + c) * 32 + lane) * 16);
#pragma unroll
    for (int nt = 0; nt < NT1; ++nt)
      acc2[nt] = __builtin_amdgcn_wmma_f32_16x16x32_f16(
          false, a, false, bf[nt], (short)0, acc2[nt], false, false);
  }

  // ---- +b2 ------------------------------------------------------------------
#pragma unroll
  for (int nt = 0; nt < NT1; ++nt) {
    const float bias = b2[nt * 16 + colb];
#pragma unroll
    for (int v = 0; v < 8; ++v) acc2[nt][v] += bias;
  }

  // ---- LayerNorm per row + epilogue ----------------------------------------
  // Row (rhalf*8+v) lives in VGPR v across the 16 lanes of this half-wave and
  // the NT1 column tiles; reduce with xor-shuffles inside the 16-lane half.
#pragma unroll
  for (int v = 0; v < 8; ++v) {
    float s = 0.f, q = 0.f;
#pragma unroll
    for (int nt = 0; nt < NT1; ++nt) {
      float t = acc2[nt][v];
      s += t;
      q += t * t;
    }
#pragma unroll
    for (int mk = 8; mk >= 1; mk >>= 1) {
      s += __shfl_xor(s, mk, 32);
      q += __shfl_xor(q, mk, 32);
    }
    const float mean = s * (1.f / DOUT);
    const float var  = q * (1.f / DOUT) - mean * mean;
    const float rstd = rsqrtf(var + 1e-5f);
    const int gr = row0 + rhalf * 8 + v;
    if (gr < nrows) {
#pragma unroll
      for (int nt = 0; nt < NT1; ++nt) {
        const int cg = nt * 16 + colb;
        float t = (acc2[nt][v] - mean) * rstd * ln_g[cg] + ln_b[cg];
        if (out) {
          float o = t;
          if (resid) {
            o += resid[(size_t)gr * DOUT + cg];
            o = o > 0.f ? o : 0.01f * o;  // LeakyReLU(0.01)
          }
          out[(size_t)gr * DOUT + cg] = o;
        }
        if (scat) atomicAdd(agg + (size_t)scat[gr] * DOUT + cg, t);
      }
    }
  }
}

// ---------------------------------------------------------------------------
// Host side
// ---------------------------------------------------------------------------
static inline void pack_w(hipStream_t s, const float* W, _Float16* dst, int K, int N) {
  int total = K * N;
  pack_w_kernel<<<(total + 255) / 256, 256, 0, s>>>(W, dst, K, N);
}

extern "C" void kernel_launch(void* const* d_in, const int* in_sizes, int n_in,
                              void* d_out, int out_size, void* d_ws, size_t ws_size,
                              hipStream_t stream) {
  (void)in_sizes; (void)n_in; (void)out_size; (void)ws_size;
  // Input flattening assumption (insertion order of setup_inputs dict and
  // nested param dicts): 0:x 1:edge_attr, then per-MLP leaves W1,b1,W2,b2,g,b
  // in order mpl1.edge, mpl1.node, mpl2.edge, mpl2.node, skip.edge, skip.node,
  // then 38:edge_index(2,E_LO) 39:g(2,E_HI) 40:m_id 41:e_idx.
  const float* x   = (const float*)d_in[0];
  const float* ea  = (const float*)d_in[1];
  auto P = [&](int i) -> const float* { return (const float*)d_in[i]; };
  const int* ei    = (const int*)d_in[38];
  const int* gfine = (const int*)d_in[39];
  const int* m_id  = (const int*)d_in[40];
  const int* e_idx = (const int*)d_in[41];
  float* outp = (float*)d_out;

  // Workspace carve (bump allocator, 256B aligned). Total ~66 MB.
  char* ws = (char*)d_ws;
  size_t off = 0;
  auto carve = [&](size_t b) -> void* {
    void* p = ws + off;
    off = (off + b + 255) & ~(size_t)255;
    return p;
  };
  // 12 packed weight matrices: (mpl1e, mpl1n, mpl2e, mpl2n, skipe, skipn) x (W1,W2)
  const int pk[12]   = {384, 64, 192, 64, 192, 128, 192, 128, 384, 128, 256, 128};
  const int pn[12]   = {64, 64, 64, 64, 128, 128, 128, 128, 128, 128, 128, 128};
  const int widx[12] = {2, 4, 8, 10, 14, 16, 20, 22, 26, 28, 32, 34};
  _Float16* pw[12];
  for (int i = 0; i < 12; ++i) pw[i] = (_Float16*)carve((size_t)pk[i] * pn[i] * 2);
  float* zrow  = (float*)carve(128 * 4);          // shared zero row for unpool
  int*   inv_n = (int*)carve((size_t)N_HI * 4);
  int*   inv_e = (int*)carve((size_t)E_HI * 4);
  float* agg   = (float*)carve((size_t)N_HI * 128 * 4);
  float* e1    = (float*)carve((size_t)E_LO * 64 * 4);
  float* x1    = (float*)carve((size_t)N_LO * 64 * 4);
  float* xsk   = (float*)carve((size_t)N_HI * 128 * 4);

  // Pack weights into WMMA B-fragment layout (reused across 480k GEMM rows).
  for (int i = 0; i < 12; ++i) pack_w(stream, P(widx[i]), pw[i], pk[i], pn[i]);

  // Zero row + inverse unpool maps (0xFF fill == -1 per int).
  hipMemsetAsync(zrow, 0, 128 * 4, stream);
  hipMemsetAsync(inv_n, 0xFF, (size_t)N_HI * 4, stream);
  hipMemsetAsync(inv_e, 0xFF, (size_t)E_HI * 4, stream);
  scatter_ids_kernel<<<(N_LO + 255) / 256, 256, 0, stream>>>(m_id, N_LO, inv_n);
  scatter_ids_kernel<<<(E_LO + 255) / 256, 256, 0, stream>>>(e_idx, E_LO, inv_e);

  dim3 blk(128);
  auto grid_for = [](int rows) { return dim3((unsigned)((rows + 63) / 64)); };

  // ---- skip branch: MPL on fine graph with unpooled x / edge_attr ----------
  hipMemsetAsync(agg, 0, (size_t)N_HI * 128 * 4, stream);
  mlp_fused<128, 128, 128, 128><<<grid_for(E_HI), blk, 0, stream>>>(
      E_HI, zrow,
      x,  gfine,        inv_n,            // xu[src] = x[inv_n[g0]]
      x,  gfine + E_HI, inv_n,            // xu[dst]
      ea, inv_e,        nullptr,          // eau     = edge_attr[inv_e[e]]
      pw[8], P(27), pw[9], P(29), P(30), P(31),
      nullptr, gfine + E_HI, agg, nullptr);
  mlp_fused<128, 128, 0, 128><<<grid_for(N_HI), blk, 0, stream>>>(
      N_HI, zrow,
      x,   inv_n,   nullptr,              // xu
      agg, nullptr, nullptr,              // scatter-sum aggregate
      nullptr, nullptr, nullptr,
      pw[10], P(33), pw[11], P(35), P(36), P(37),
      xsk, nullptr, nullptr, nullptr);

  // ---- mpl1: coarse graph --------------------------------------------------
  hipMemsetAsync(agg, 0, (size_t)N_LO * 64 * 4, stream);
  mlp_fused<128, 128, 128, 64><<<grid_for(E_LO), blk, 0, stream>>>(
      E_LO, zrow,
      x,  ei,        nullptr,             // x[src]
      x,  ei + E_LO, nullptr,             // x[dst]
      ea, nullptr,   nullptr,             // edge_attr
      pw[0], P(3), pw[1], P(5), P(6), P(7),
      e1, ei + E_LO, agg, nullptr);       // keep e1 for mpl2; scatter-sum
  mlp_fused<128, 64, 0, 64><<<grid_for(N_LO), blk, 0, stream>>>(
      N_LO, zrow,
      x,   nullptr, nullptr,
      agg, nullptr, nullptr,
      nullptr, nullptr, nullptr,
      pw[2], P(9), pw[3], P(11), P(12), P(13),
      x1, nullptr, nullptr, nullptr);

  // ---- mpl2: fine graph with unpooled x1 / e1 ------------------------------
  hipMemsetAsync(agg, 0, (size_t)N_HI * 128 * 4, stream);
  mlp_fused<64, 64, 64, 128><<<grid_for(E_HI), blk, 0, stream>>>(
      E_HI, zrow,
      x1, gfine,        inv_n,            // xu1[src]
      x1, gfine + E_HI, inv_n,            // xu1[dst]
      e1, inv_e,        nullptr,          // eu1
      pw[4], P(15), pw[5], P(17), P(18), P(19),
      nullptr, gfine + E_HI, agg, nullptr);
  mlp_fused<64, 128, 0, 128><<<grid_for(N_HI), blk, 0, stream>>>(
      N_HI, zrow,
      x1,  inv_n,   nullptr,              // xu1
      agg, nullptr, nullptr,
      nullptr, nullptr, nullptr,
      pw[6], P(21), pw[7], P(23), P(24), P(25),
      outp, nullptr, nullptr, xsk);       // out = leaky_relu(x2 + x_skip)
}
